// TimeSeriesLSTMRegressor_74062416052864
// MI455X (gfx1250) — matrix operand order repaired and checked by reading, not compiled, not run on hardware
//
#include <hip/hip_runtime.h>
#include <hip/hip_bf16.h>
#include <stdint.h>

typedef __attribute__((ext_vector_type(16))) __bf16 v16bf;
typedef __attribute__((ext_vector_type(8)))  float  v8f;
typedef __attribute__((ext_vector_type(4)))  unsigned int v4u;

union Frag16 { v16bf v; v4u q[2]; };

__device__ __forceinline__ __bf16 to_bf16(float f) {
  union { float f; uint32_t u; } in; in.f = f;
  uint32_t u = (in.u + 0x7FFFu + ((in.u >> 16) & 1u)) >> 16;
  union { uint16_t s; __bf16 b; } out; out.s = (uint16_t)u;
  return out.b;
}

__device__ __forceinline__ float sigmoidf_(float x) { return 1.0f / (1.0f + __expf(-x)); }

// ---------------- fp32 -> bf16 bulk convert ----------------
__global__ void cvt_f32_to_bf16(const float* __restrict__ src, __bf16* __restrict__ dst, int n) {
  for (int i = blockIdx.x * blockDim.x + threadIdx.x; i < n; i += gridDim.x * blockDim.x)
    dst[i] = to_bf16(src[i]);
}

// ---------------- W[N][K] fp32 -> WT[K][N] bf16 (row-major transpose, for LDS staging) ----
__global__ void transpose_cvt(const float* __restrict__ W, __bf16* __restrict__ WT, int N, int K) {
  int total = N * K;
  for (int i = blockIdx.x * blockDim.x + threadIdx.x; i < total; i += gridDim.x * blockDim.x) {
    int n = i % N, k = i / N;
    WT[i] = to_bf16(W[(size_t)n * K + k]);
  }
}

// ---------------- W[N][K] fp32 -> fragment-major packed bf16 ----------------
// P[((kb*(N/16) + nt)*32 + lane)*16 + j] = W[nt*16 + lane%16][kb*32 + kt(lane/16, j)], kt = table k order
// -> a WMMA B-fragment is one fully coalesced 1KB access.
__global__ void pack_b(const float* __restrict__ W, __bf16* __restrict__ P, int N, int K) {
  int total = N * K;
  int nt16 = N >> 4;
  for (int i = blockIdx.x * blockDim.x + threadIdx.x; i < total; i += gridDim.x * blockDim.x) {
    int j    = i & 15;
    int lane = (i >> 4) & 31;
    int blk  = i >> 9;             // kb*(N/16) + nt
    int nt   = blk % nt16, kb = blk / nt16;
    int kk = (j < 8) ? 8 * (lane >> 4) + j : 8 + 8 * (lane >> 4) + j;
    P[i] = to_bf16(W[(size_t)(nt * 16 + (lane & 15)) * K + kb * 32 + kk]);
  }
}

// ---------------- C[M][N] = A[M][K](bf16) @ B(packed bf16) + bih[N] + bhh[N] ----------------
// Wave tile 16(M) x 64(N), K-step 32. Software pipeline uses ONLY statically-indexed named
// buffers (a0/a1, b0[]/b1[] with unrolled constant indices) + manual x2 K-unroll, so the
// fragments stay in VGPRs (a loop-carried 'buf' index would become an alloca -> LDS/scratch).
template <int N, int K>
__global__ __launch_bounds__(256)
void gemm_xg_wmma(const __bf16* __restrict__ A, const __bf16* __restrict__ BP,
                  const float* __restrict__ bih, const float* __restrict__ bhh,
                  float* __restrict__ C, int M) {
  const int wave  = threadIdx.x >> 5;
  const int lane  = threadIdx.x & 31;
  const int half  = lane >> 4;
  const int nlane = lane & 15;
  const int w     = blockIdx.x * 8 + wave;
  constexpr int numNT = N >> 6;
  constexpr int NT16  = N >> 4;
  const int tiles = (M >> 4) * numNT;
  if (w >= tiles) return;
  const int m0 = (w / numNT) << 4;
  const int n0 = (w % numNT) << 6;

  v8f acc[4];
#pragma unroll
  for (int t = 0; t < 4; ++t) {
    int n = n0 + 16 * t + nlane;
    float b = bih[n] + bhh[n];
#pragma unroll
    for (int r = 0; r < 8; ++r) acc[t][r] = b;
  }

  const __bf16* arow = A + (size_t)(m0 + nlane) * K;

  Frag16 a0, a1;
  Frag16 b0[4], b1[4];
  auto loadA = [&](Frag16& a, int k0) {
    a.q[0] = *(const v4u*)(arow + k0 + 8 * half);
    a.q[1] = *(const v4u*)(arow + k0 + 16 + 8 * half);
  };
  auto loadB = [&](Frag16* b, int k0) {
    const __bf16* bk = BP + (size_t)(((k0 >> 5) * NT16 + (n0 >> 4)) * 32 + lane) * 16;
#pragma unroll
    for (int t = 0; t < 4; ++t) {
      b[t].q[0] = *(const v4u*)(bk + t * 512);       // nt+1 -> +32*16 elements
      b[t].q[1] = *(const v4u*)(bk + t * 512 + 8);
    }
  };
  auto mm = [&](const Frag16& a, const Frag16* b) {
#pragma unroll
    for (int t = 0; t < 4; ++t)
      acc[t] = __builtin_amdgcn_wmma_f32_16x16x32_bf16(false, a.v, false, b[t].v,
                                                       (short)0, acc[t], false, false);
  };

  loadA(a0, 0);
  loadB(b0, 0);
  for (int k0 = 0; k0 < K; k0 += 64) {      // K % 64 == 0 for all instantiations
    loadA(a1, k0 + 32);
    loadB(b1, k0 + 32);
    mm(a0, b0);
    if (k0 + 64 < K) {
      loadA(a0, k0 + 64);
      loadB(b0, k0 + 64);
    }
    mm(a1, b1);
  }

#pragma unroll
  for (int t = 0; t < 4; ++t)
#pragma unroll
    for (int r = 0; r < 8; ++r)
      C[(size_t)(m0 + r + 8 * half) * N + n0 + 16 * t + nlane] = acc[t][r];
}

// ---------------- LSTM recurrence ----------------
// 4 WGs, one per 16-row batch slice (independent recurrences; batch is the parallel axis).
// One wave per 16 hidden units, all four gates -> i/f/g/o share lane/VGPR slots across 4
// accumulators; c stays in registers; h double-buffered in padded LDS rows.
// xg for step t+1 is prefetched via global_prefetch (zero VGPR cost); the elementwise phase
// loads xg directly and hits cache.
// WLDS: Whh^T (row-major) staged in dynamic LDS with padded rows (layers 2/3).
// Layer 1 streams fragment-major packed Whh through a 4-deep register ring (fully unrolled,
// constant indices -> register-resident; coalesced 1KB fragment loads).
template <int H, bool WLDS, int NTHREADS>
__global__ __launch_bounds__(NTHREADS, 1)
void lstm_rec(const float* __restrict__ xg,     // [B][T][4H] (already + bih + bhh)
              const __bf16* __restrict__ Whh,   // WLDS: row-major WT[H][4H]; else packed
              __bf16* __restrict__ out,         // [B][T][H]
              float* __restrict__ hlast,        // [B][H]
              int T) {
  constexpr int N4  = 4 * H;
  constexpr int NW  = H / 16;    // active waves (16 / 8 / 4)
  constexpr int KS  = H / 32;    // WMMA K-steps
  constexpr int NFR = KS * 4;    // flattened WMMA count per step
  constexpr int HP  = H + 8;     // padded h row (stride % 64 banks != 0, keeps 16B align)
  constexpr int WP  = N4 + 8;    // padded w row

  extern __shared__ __bf16 smem[];
  __bf16* h_lds = smem;                       // [2][16][HP]
  __bf16* w_lds = smem + 2 * 16 * HP;         // [H][WP] when WLDS

  const int tid   = threadIdx.x;
  const int wave  = tid >> 5;
  const int lane  = tid & 31;
  const int half  = lane >> 4;
  const int nlane = lane & 15;
  const int b0    = blockIdx.x * 16;
  const bool active = (wave < NW);
  const int u0    = wave * 16;

  for (int i = tid; i < 2 * 16 * HP; i += NTHREADS) h_lds[i] = to_bf16(0.0f);
  if (WLDS)
    for (int i = tid; i < H * N4; i += NTHREADS) {
      int k = i / N4, n = i % N4;
      int L = k & 31, s = n & 15;
      int kk = (k & ~31) + ((s < 8) ? 8 * (L >> 4) + s : 8 + 8 * (L >> 4) + s);
      w_lds[k * WP + n] = Whh[kk * N4 + (n & ~15) + (L & 15)];
    }
  __syncthreads();

  // layer-1 packed Whh fragment address: frag (ks, gi) at nt = gi*(H/16) + wave
  auto bgptr = [&](int idx) -> const __bf16* {
    int ks = idx >> 2, gi = idx & 3;
    return Whh + (size_t)((ks * (N4 >> 4) + gi * (H >> 4) + wave) * 32 + lane) * 16;
  };
  // prefetch one (row, 2 gate-regions) worth of next-step xg per lane: zero VGPR footprint
  auto xgprefetch = [&](int t) {
    int r  = lane & 15;   // batch row m
    int gi = lane >> 4;   // 0..1; +2 covers all four gates across the two halves
    const float* base = xg + (size_t)((b0 + r) * T + t) * N4 + u0;
    __builtin_prefetch(base + gi * H, 0, 3);
    __builtin_prefetch(base + (gi + 2) * H, 0, 3);
  };

  v8f c;
#pragma unroll
  for (int r = 0; r < 8; ++r) c[r] = 0.0f;

  if (active) xgprefetch(0);

  int cur = 0;
  for (int t = 0; t < T; ++t) {
    if (active) {
      if (t + 1 < T) xgprefetch(t + 1);  // full step of prefetch distance

      v8f acc[4];
#pragma unroll
      for (int gi = 0; gi < 4; ++gi)
#pragma unroll
        for (int r = 0; r < 8; ++r) acc[gi][r] = 0.0f;

      const __bf16* hbase = h_lds + cur * 16 * HP + nlane * HP;

      if (WLDS) {
#pragma unroll
        for (int ks = 0; ks < KS; ++ks) {
          Frag16 a;
          a.q[0] = *(const v4u*)(hbase + ks * 32 + 8 * half);
          a.q[1] = *(const v4u*)(hbase + ks * 32 + 16 + 8 * half);
          const __bf16* wrow = w_lds + (size_t)(ks * 32 + lane) * WP + u0;
#pragma unroll
          for (int gi = 0; gi < 4; ++gi) {
            Frag16 b;
            b.q[0] = *(const v4u*)(wrow + gi * H);
            b.q[1] = *(const v4u*)(wrow + gi * H + 8);
            acc[gi] = __builtin_amdgcn_wmma_f32_16x16x32_bf16(
                false, a.v, false, b.v, (short)0, acc[gi], false, false);
          }
        }
      } else {
        // layer 1: stream packed Whh from L2 through a 4-deep register ring (prefetch 4 ahead)
        Frag16 bring[4];
#pragma unroll
        for (int p = 0; p < 4; ++p) {
          const __bf16* bp = bgptr(p);
          bring[p].q[0] = *(const v4u*)(bp);
          bring[p].q[1] = *(const v4u*)(bp + 8);
        }
        Frag16 a;
#pragma unroll
        for (int idx = 0; idx < NFR; ++idx) {
          int ks = idx >> 2, gi = idx & 3;
          if (gi == 0) {
            a.q[0] = *(const v4u*)(hbase + ks * 32 + 8 * half);
            a.q[1] = *(const v4u*)(hbase + ks * 32 + 16 + 8 * half);
          }
          acc[gi] = __builtin_amdgcn_wmma_f32_16x16x32_bf16(
              false, a.v, false, bring[idx & 3].v, (short)0, acc[gi], false, false);
          if (idx + 4 < NFR) {
            const __bf16* bp = bgptr(idx + 4);
            bring[idx & 3].q[0] = *(const v4u*)(bp);
            bring[idx & 3].q[1] = *(const v4u*)(bp + 8);
          }
        }
      }

      // cell update entirely in registers; xg loads hit cache (prefetched last step)
      int nxt = cur ^ 1;
      int u = u0 + nlane;
      const float* xrow = xg + (size_t)t * N4 + u0 + nlane;
#pragma unroll
      for (int r = 0; r < 8; ++r) {
        int m = r + 8 * half;
        const float* xv = xrow + (size_t)(b0 + m) * T * N4;
        float ig = sigmoidf_(acc[0][r] + xv[0 * H]);
        float fg = sigmoidf_(acc[1][r] + xv[1 * H]);
        float gg = tanhf(acc[2][r] + xv[2 * H]);
        float og = sigmoidf_(acc[3][r] + xv[3 * H]);
        float cc = fg * c[r] + ig * gg;
        c[r] = cc;
        float hh = og * tanhf(cc);
        __bf16 hb = to_bf16(hh);
        h_lds[nxt * 16 * HP + m * HP + u] = hb;
        out[(size_t)((b0 + m) * T + t) * H + u] = hb;
        if (t == T - 1) hlast[(b0 + m) * H + u] = hh;
      }
    }
    __syncthreads();
    cur ^= 1;
  }
}

// ---------------- final deep+wide head ----------------
__global__ void head_kernel(const float* __restrict__ x, const float* __restrict__ h3,
                            const float* __restrict__ Wd, const float* __restrict__ bd,
                            const float* __restrict__ Ww, const float* __restrict__ bw,
                            const float* __restrict__ Wo, const float* __restrict__ bo,
                            float* __restrict__ out, int T) {
  int b = threadIdx.x;
  if (b >= 64) return;
  const float* xl = x + ((size_t)b * T + (T - 1)) * 2048;
  const float* hd = h3 + b * 64;
  float res = bo[0];
  for (int j = 0; j < 32; ++j) {
    float s = bd[j];
    for (int i = 0; i < 64; ++i) s += hd[i] * Wd[j * 64 + i];
    res += fmaxf(s, 0.0f) * Wo[j];
  }
  for (int j = 0; j < 32; ++j) {
    float s = bw[j];
    for (int i = 0; i < 2048; ++i) s += xl[i] * Ww[j * 2048 + i];
    res += fmaxf(s, 0.0f) * Wo[32 + j];
  }
  out[b] = res;
}

extern "C" void kernel_launch(void* const* d_in, const int* in_sizes, int n_in,
                              void* d_out, int out_size, void* d_ws, size_t ws_size,
                              hipStream_t stream) {
  const float* x    = (const float*)d_in[0];
  const float* Wih1 = (const float*)d_in[1];
  const float* Whh1 = (const float*)d_in[2];
  const float* bih1 = (const float*)d_in[3];
  const float* bhh1 = (const float*)d_in[4];
  const float* Wih2 = (const float*)d_in[5];
  const float* Whh2 = (const float*)d_in[6];
  const float* bih2 = (const float*)d_in[7];
  const float* bhh2 = (const float*)d_in[8];
  const float* Wih3 = (const float*)d_in[9];
  const float* Whh3 = (const float*)d_in[10];
  const float* bih3 = (const float*)d_in[11];
  const float* bhh3 = (const float*)d_in[12];
  const float* Wd   = (const float*)d_in[13];
  const float* bd   = (const float*)d_in[14];
  const float* Ww   = (const float*)d_in[15];
  const float* bw   = (const float*)d_in[16];
  const float* Wo   = (const float*)d_in[17];
  const float* bo   = (const float*)d_in[18];

  const int B = 64, T = 512, D = 2048;
  char* ws = (char*)d_ws;
  size_t off = 0;
  auto take = [&](size_t bytes) -> char* {
    char* p = ws + off;
    off += (bytes + 255) & ~(size_t)255;
    return p;
  };

  __bf16* xbf   = (__bf16*)take((size_t)B * T * D * 2);        // 128 MB (L2-resident)
  __bf16* WihP1 = (__bf16*)take((size_t)2048 * 1024 * 2);      // packed (fragment-major)
  __bf16* WhhP1 = (__bf16*)take((size_t)256 * 1024 * 2);       // packed (fragment-major)
  __bf16* WihP2 = (__bf16*)take((size_t)256 * 512 * 2);        // packed
  __bf16* WhhT2 = (__bf16*)take((size_t)128 * 512 * 2);        // row-major transpose
  __bf16* WihP3 = (__bf16*)take((size_t)128 * 256 * 2);        // packed
  __bf16* WhhT3 = (__bf16*)take((size_t)64 * 256 * 2);         // row-major transpose
  float*  xg    = (float*)take((size_t)B * T * 1024 * 4);      // 128 MB, reused per layer
  __bf16* out1  = (__bf16*)take((size_t)B * T * 256 * 2);
  __bf16* out2  = (__bf16*)take((size_t)B * T * 128 * 2);
  __bf16* out3  = (__bf16*)take((size_t)B * T * 64 * 2);
  float*  hlast = (float*)take((size_t)B * 256 * 4);

  // precision conversion + weight packing
  cvt_f32_to_bf16<<<4096, 256, 0, stream>>>(x, xbf, B * T * D);
  pack_b<<<2048, 256, 0, stream>>>(Wih1, WihP1, 1024, 2048);
  pack_b<<<256, 256, 0, stream>>>(Whh1, WhhP1, 1024, 256);
  pack_b<<<128, 256, 0, stream>>>(Wih2, WihP2, 512, 256);
  transpose_cvt<<<64, 256, 0, stream>>>(Whh2, WhhT2, 512, 128);
  pack_b<<<32, 256, 0, stream>>>(Wih3, WihP3, 256, 128);
  transpose_cvt<<<16, 256, 0, stream>>>(Whh3, WhhT3, 256, 64);

  const int M = B * T;  // 32768

  // dynamic LDS sizes (padded rows)
  const int smem1 = 2 * 16 * (256 + 8) * 2;                          // 16.5 KB
  const int smem2 = 2 * 16 * (128 + 8) * 2 + 128 * (512 + 8) * 2;   // ~139 KB
  const int smem3 = 2 * 16 * (64 + 8) * 2 + 64 * (256 + 8) * 2;     // ~37.5 KB

  // layer 1: 16 waves/WG, packed Whh streamed from L2
  gemm_xg_wmma<1024, 2048><<<4096, 256, 0, stream>>>(xbf, WihP1, bih1, bhh1, xg, M);
  lstm_rec<256, false, 512><<<4, 512, smem1, stream>>>(xg, WhhP1, out1, hlast, T);
  // layer 2: 8 waves/WG, Whh^T resident in LDS
  gemm_xg_wmma<512, 256><<<2048, 256, 0, stream>>>(out1, WihP2, bih2, bhh2, xg, M);
  lstm_rec<128, true, 256><<<4, 256, smem2, stream>>>(xg, WhhT2, out2, hlast, T);
  // layer 3: 4 waves/WG, Whh^T resident in LDS
  gemm_xg_wmma<256, 128><<<1024, 256, 0, stream>>>(out2, WihP3, bih3, bhh3, xg, M);
  lstm_rec<64, true, 128><<<4, 128, smem3, stream>>>(xg, WhhT3, out3, hlast, T);

  // heads
  head_kernel<<<1, 64, 0, stream>>>(x, hlast, Wd, bd, Ww, bw, Wo, bo, (float*)d_out, T);
}
